// SpanBERTCorefModel_14955076125127
// MI455X (gfx1250) — matrix operand order repaired and checked by reading, not compiled, not run on hardware
//
#include <hip/hip_runtime.h>

// ---------------- problem constants (match reference) ----------------
#define LSEG 8
#define NCAND 7168
#define SLEN 256
#define MAXW 30
#define KOUT 128
#define DEMB 2304
#define DHID 150
#define PAD  (SLEN + MAXW + 1)      // 287
#define NT   10                     // ceil(150/16) n-tiles of hidden dim
#define KC   (DEMB / 32)            // 72 k-chunks of 32
#define ROWS (LSEG * NCAND)         // 57344
#define SORTN 8192                  // next pow2 >= NCAND

// frag buffer: KC * NT * 2(split) * 32(lane) * 8 uints (= 16 bf16 per lane)
#define FRAG_UINTS (KC * NT * 2 * 32 * 8)
#define FRAG_BYTES (FRAG_UINTS * 4)          // 1,474,560
#define SCORE_OFF  FRAG_BYTES                // scores: ROWS floats
#define KEYS_OFF   (SCORE_OFF + ROWS * 4)    // keys: LSEG * SORTN u64

typedef __attribute__((ext_vector_type(16))) __bf16          v16bf;
typedef __attribute__((ext_vector_type(16))) unsigned short  v16us;
typedef __attribute__((ext_vector_type(8)))  float           v8f;
typedef __attribute__((ext_vector_type(8)))  unsigned int    v8u;

__device__ __forceinline__ unsigned short bf16_rne(float f) {
  union { float f; unsigned u; } cv; cv.f = f;
  unsigned u = cv.u;
  return (unsigned short)((u + 0x7FFFu + ((u >> 16) & 1u)) >> 16);
}
__device__ __forceinline__ float bf16_f32(unsigned short h) {
  union { unsigned u; float f; } cv; cv.u = ((unsigned)h) << 16;
  return cv.f;
}

// ---------------------------------------------------------------------
// Kernel 0: pre-swizzle W1 into split-bf16 WMMA B-fragments.
// B 32x16 bf16 frag, contiguous k order (same as A in score_gemm): lane = ncol + 16*(k>=16),
// vgpr r holds K=2r,2r+1 (low half = even k). frag linear index
// ((kc*NT+nt)*2+split)*32+lane, 8 uints each.
// ---------------------------------------------------------------------
__global__ __launch_bounds__(256) void prep_w1_frags(
    const float* __restrict__ W1, unsigned int* __restrict__ frags) {
  int t = blockIdx.x * 256 + threadIdx.x;
  if (t >= KC * NT * 2 * 32) return;
  int lane  = t & 31;
  int split = (t >> 5) & 1;
  int nt    = (t >> 6) % NT;
  int kc    = t / (NT * 2 * 32);
  int ncol  = nt * 16 + (lane & 15);
  int kbase = kc * 32 + (lane >> 4) * 16;
  unsigned int out[8];
#pragma unroll
  for (int jj = 0; jj < 8; ++jj) {
    unsigned short h[2];
#pragma unroll
    for (int q = 0; q < 2; ++q) {
      int k = kbase + 2 * jj + q;
      float w = (ncol < DHID) ? W1[(size_t)k * DHID + ncol] : 0.0f;
      unsigned short hi = bf16_rne(w);
      h[q] = (split == 0) ? hi : bf16_rne(w - bf16_f32(hi));
    }
    out[jj] = (unsigned int)h[0] | ((unsigned int)h[1] << 16);
  }
  unsigned int* dst = frags + (size_t)t * 8;
#pragma unroll
  for (int jj = 0; jj < 8; ++jj) dst[jj] = out[jj];
}

// ---------------------------------------------------------------------
// Kernel 1: split-bf16 WMMA GEMM + bias + ReLU + dot(W2) -> scores[ROWS].
// 256 threads (8 waves). Wave w: rows block*128 + w*16 .. +15.
// A 16x32 bf16, contiguous k: lane = m + 16*(k>>4); vgpr = (k&15)>>1.
// ---------------------------------------------------------------------
__global__ __launch_bounds__(256) void score_gemm(
    const float* __restrict__ emb, const float* __restrict__ b1,
    const float* __restrict__ W2, const float* __restrict__ b2,
    const unsigned int* __restrict__ frags, float* __restrict__ scores) {
  __shared__ __align__(32) unsigned int ldsB[NT * 2 * 32 * 8];  // 20 KB
  const int wave = threadIdx.x >> 5;
  const int lane = threadIdx.x & 31;
  const int rowBase = blockIdx.x * 128 + wave * 16;
  const int myRow = rowBase + (lane & 15);
  const float* rowPtr = emb + (size_t)myRow * DEMB;
  const int c0 = (lane >> 4) * 16;  // contiguous 16-k run for this lane

  v8f acc[NT];
#pragma unroll
  for (int t = 0; t < NT; ++t)
#pragma unroll
    for (int r = 0; r < 8; ++r) acc[t][r] = 0.0f;

  for (int kc = 0; kc < KC; ++kc) {
    __syncthreads();
    {  // cooperative stage of this k-chunk's B fragments into LDS
      const uint4* s4 = (const uint4*)(frags + (size_t)kc * (NT * 2 * 32 * 8));
      uint4* d4 = (uint4*)ldsB;
#pragma unroll
      for (int i = 0; i < 5; ++i) d4[threadIdx.x + 256 * i] = s4[threadIdx.x + 256 * i];
    }
    __syncthreads();

    // A: one contiguous 16-float run per lane; split fp32 -> bf16 hi/lo
    const float* p = rowPtr + kc * 32 + c0;
    const float4 t0 = *(const float4*)(p + 0);
    const float4 t1 = *(const float4*)(p + 4);
    const float4 t2 = *(const float4*)(p + 8);
    const float4 t3 = *(const float4*)(p + 12);
    float f[16] = {t0.x, t0.y, t0.z, t0.w, t1.x, t1.y, t1.z, t1.w,
                   t2.x, t2.y, t2.z, t2.w, t3.x, t3.y, t3.z, t3.w};
    v16us ah, al;
#pragma unroll
    for (int j = 0; j < 16; ++j) {
      unsigned short hb = bf16_rne(f[j]);
      ah[j] = hb;
      al[j] = bf16_rne(f[j] - bf16_f32(hb));
    }
    v16bf ahi = __builtin_bit_cast(v16bf, ah);
    v16bf alo = __builtin_bit_cast(v16bf, al);

#pragma unroll
    for (int nt = 0; nt < NT; ++nt) {
      v16bf bhi = __builtin_bit_cast(
          v16bf, *(const v8u*)&ldsB[((nt * 2 + 0) * 32 + lane) * 8]);
      v16bf blo = __builtin_bit_cast(
          v16bf, *(const v8u*)&ldsB[((nt * 2 + 1) * 32 + lane) * 8]);
      acc[nt] = __builtin_amdgcn_wmma_f32_16x16x32_bf16(
          false, ahi, false, bhi, (short)0, acc[nt], false, false);
      acc[nt] = __builtin_amdgcn_wmma_f32_16x16x32_bf16(
          false, alo, false, bhi, (short)0, acc[nt], false, false);
      acc[nt] = __builtin_amdgcn_wmma_f32_16x16x32_bf16(
          false, ahi, false, blo, (short)0, acc[nt], false, false);
    }
  }

  // epilogue: h=relu(acc+b1); score_row = sum_n h*W2[n]  (+b2)
  float ps[8];
#pragma unroll
  for (int r = 0; r < 8; ++r) ps[r] = 0.0f;
  const float b2v = b2[0];
#pragma unroll
  for (int nt = 0; nt < NT; ++nt) {
    int n = nt * 16 + (lane & 15);
    float b1n = (n < DHID) ? b1[n] : 0.0f;
    float w2n = (n < DHID) ? W2[n] : 0.0f;
#pragma unroll
    for (int r = 0; r < 8; ++r) {
      float h = acc[nt][r] + b1n;
      h = h > 0.0f ? h : 0.0f;
      ps[r] += h * w2n;
    }
  }
#pragma unroll
  for (int m = 1; m < 16; m <<= 1)
#pragma unroll
    for (int r = 0; r < 8; ++r) ps[r] += __shfl_xor(ps[r], m, 32);
  if (lane == 0)
#pragma unroll
    for (int r = 0; r < 8; ++r) scores[rowBase + r] = ps[r] + b2v;
  if (lane == 16)
#pragma unroll
    for (int r = 0; r < 8; ++r) scores[rowBase + 8 + r] = ps[r] + b2v;
}

// ---------------------------------------------------------------------
// Kernel 2: per-segment stable sort (desc score) + greedy non-crossing
// selection + stable (start,end) sort of the K winners.
// One block (1024 thr) per segment. Keys live in workspace.
// ---------------------------------------------------------------------
__global__ __launch_bounds__(1024) void select_spans(
    const float* __restrict__ scores, const int* __restrict__ cstarts,
    const int* __restrict__ cends, unsigned long long* __restrict__ keys_ws,
    float* __restrict__ out_scores, int* __restrict__ out_idx) {
  const int seg = blockIdx.x;
  unsigned long long* keys = keys_ws + (size_t)seg * SORTN;
  const int* st = cstarts + seg * NCAND;
  const int* en = cends + seg * NCAND;

  __shared__ int le[PAD];
  __shared__ int es[PAD];
  __shared__ int sel[KOUT];
  __shared__ int selkey[KOUT];
  __shared__ int selval[KOUT];
  __shared__ int n_sel_sh;

  // build sort keys: ascending u64 == descending score, ties by index (stable)
  for (int i = threadIdx.x; i < SORTN; i += 1024) {
    unsigned long long kk;
    if (i < NCAND) {
      union { float f; unsigned u; } cv; cv.f = scores[seg * NCAND + i];
      unsigned m = (cv.u >> 31) ? ~cv.u : (cv.u | 0x80000000u);
      kk = (((unsigned long long)(~m)) << 32) | (unsigned)i;
    } else {
      kk = 0xFFFFFFFFFFFFFFFFull;
    }
    keys[i] = kk;
  }
  for (int i = threadIdx.x; i < PAD; i += 1024) { le[i] = -1; es[i] = PAD; }
  for (int i = threadIdx.x; i < KOUT; i += 1024) sel[i] = 0;
  __syncthreads();

  // bitonic sort ascending (8192 u64)
  for (int k = 2; k <= SORTN; k <<= 1) {
    for (int j = k >> 1; j > 0; j >>= 1) {
      for (int i = threadIdx.x; i < SORTN; i += 1024) {
        int p = i ^ j;
        if (p > i) {
          bool up = ((i & k) == 0);
          unsigned long long a = keys[i], b = keys[p];
          if ((a > b) == up) { keys[i] = b; keys[p] = a; }
        }
      }
      __syncthreads();
    }
  }

  // sequential greedy scan by wave 0; 31-wide crossing window across lanes.
  if (threadIdx.x < 32) {
    const int lane = threadIdx.x;
    volatile int* vle = le;
    volatile int* ves = es;
    int n_sel = 0;
    for (int base = 0; base < NCAND && n_sel < KOUT; base += 32) {
      int ig = base + lane;
      int cand = (int)(keys[ig] & 0xFFFFFFFFu);
      int sL = 0, eL = 0;
      if (ig < NCAND) { sL = st[cand]; eL = en[cand]; }
      int nIter = (NCAND - base < 32) ? (NCAND - base) : 32;
      for (int t = 0; t < nIter; ++t) {
        int s = __shfl(sL, t, 32);
        int e = __shfl(eL, t, 32);
        int cd = __shfl(cand, t, 32);
        int w = e - s;
        int lev = vle[s + lane];
        int esv = ves[s + lane];
        bool cross = ((lane >= 1) && (lane <= w) && (lev > e)) ||
                     ((lane < w) && (esv < s));
        if (__ballot(cross) == 0ull) {
          if (lane == 0) {
            sel[n_sel] = cd;
            int cur = vle[s]; vle[s] = (cur > e) ? cur : e;
            int ce = ves[e];  ves[e] = (ce < s) ? ce : s;
          }
          asm volatile("s_wait_dscnt 0" ::: "memory");
          n_sel++;
          if (n_sel >= KOUT) break;
        }
      }
    }
    if (lane == 0) n_sel_sh = n_sel;
  }
  __syncthreads();

  // stable sort of selected by (start*PAD+end), ties by selection slot
  if (threadIdx.x < KOUT) {
    int slot = threadIdx.x, n = n_sel_sh, v = sel[slot];
    selkey[slot] = (slot < n) ? (st[v] * PAD + en[v]) * 128 + slot
                              : 0x7F000000 + slot;
    selval[slot] = v;
  }
  __syncthreads();
  for (int k = 2; k <= KOUT; k <<= 1) {
    for (int j = k >> 1; j > 0; j >>= 1) {
      if (threadIdx.x < KOUT) {
        int i = threadIdx.x, p = i ^ j;
        if (p > i) {
          bool up = ((i & k) == 0);
          int a = selkey[i], b = selkey[p];
          if ((a > b) == up) {
            selkey[i] = b; selkey[p] = a;
            int va = selval[i]; selval[i] = selval[p]; selval[p] = va;
          }
        }
      }
      __syncthreads();
    }
  }
  if (threadIdx.x < KOUT) {
    int slot = threadIdx.x, n = n_sel_sh;
    int outI = (slot < n) ? selval[slot] : selval[0];
    out_scores[seg * KOUT + slot] = scores[seg * NCAND + outI];
    out_idx[seg * KOUT + slot] = outI;
  }
}

// ---------------------------------------------------------------------
extern "C" void kernel_launch(void* const* d_in, const int* in_sizes, int n_in,
                              void* d_out, int out_size, void* d_ws,
                              size_t ws_size, hipStream_t stream) {
  const float* emb    = (const float*)d_in[0];
  const float* W1     = (const float*)d_in[1];
  const float* b1     = (const float*)d_in[2];
  const float* W2     = (const float*)d_in[3];
  const float* b2     = (const float*)d_in[4];
  const int*   starts = (const int*)d_in[5];
  const int*   ends   = (const int*)d_in[6];
  (void)in_sizes; (void)n_in; (void)out_size; (void)ws_size;

  float* out_scores = (float*)d_out;
  int*   out_idx    = (int*)((float*)d_out + LSEG * KOUT);

  unsigned int* frags = (unsigned int*)d_ws;
  float* scores = (float*)((char*)d_ws + SCORE_OFF);
  unsigned long long* keys = (unsigned long long*)((char*)d_ws + KEYS_OFF);

  prep_w1_frags<<<(KC * NT * 2 * 32 + 255) / 256, 256, 0, stream>>>(W1, frags);
  score_gemm<<<ROWS / 128, 256, 0, stream>>>(emb, b1, W2, b2, frags, scores);
  select_spans<<<LSEG, 1024, 0, stream>>>(scores, starts, ends, keys,
                                          out_scores, out_idx);
}
